// PositionalEncoding_83167746720502
// MI455X (gfx1250) — compile-verified, matrix operand order checked by reading, not run on hardware
//
#include <hip/hip_runtime.h>

// ---- problem constants (match reference) ----
#define DD      512
#define LSEQ    4096
#define BATCH   8
#define M_TOT   (LSEQ * BATCH)   // 32768
#define LN_EPS  1e-5f

typedef __bf16 bf16_t;
typedef bf16_t bf16x4 __attribute__((ext_vector_type(4)));
typedef bf16_t bf16x8 __attribute__((ext_vector_type(8)));
typedef bf16_t v16bf  __attribute__((ext_vector_type(16)));
typedef float  v8f    __attribute__((ext_vector_type(8)));
typedef unsigned int u32x4 __attribute__((ext_vector_type(4)));
typedef int          i32x4 __attribute__((ext_vector_type(4)));
typedef int          i32x8 __attribute__((ext_vector_type(8)));

// ---------------- fp32 -> bf16 conversion (vectorized) ----------------
__global__ __launch_bounds__(256) void k_cvt(const float* __restrict__ s,
                                             bf16_t* __restrict__ d, int n4) {
    int i = blockIdx.x * 256 + threadIdx.x;
    if (i >= n4) return;
    float4 v = ((const float4*)s)[i];
    bf16x4 o = { (bf16_t)v.x, (bf16_t)v.y, (bf16_t)v.z, (bf16_t)v.w };
    ((bf16x4*)d)[i] = o;
}

// split W (512 x 1024, row-major) into Wx (cols 0..511) and Wpe (cols 512..1023), bf16
__global__ __launch_bounds__(256) void k_cvt_w(const float* __restrict__ W,
                                               bf16_t* __restrict__ Wxb,
                                               bf16_t* __restrict__ Wpeb) {
    int idx = (blockIdx.x * 256 + threadIdx.x) * 4;   // over 512*512 elements
    if (idx >= DD * DD) return;
    int dch = idx >> 9, k = idx & 511;
    float4 vx = *(const float4*)(W + (size_t)dch * 1024 + k);
    float4 vp = *(const float4*)(W + (size_t)dch * 1024 + 512 + k);
    bf16x4 ox = { (bf16_t)vx.x, (bf16_t)vx.y, (bf16_t)vx.z, (bf16_t)vx.w };
    bf16x4 op = { (bf16_t)vp.x, (bf16_t)vp.y, (bf16_t)vp.z, (bf16_t)vp.w };
    *(bf16x4*)(Wxb  + (size_t)dch * DD + k) = ox;
    *(bf16x4*)(Wpeb + (size_t)dch * DD + k) = op;
}

// ---------------- uniformize a wave-uniform value into SGPRs ----------------
__device__ __forceinline__ unsigned long long rfl64(unsigned long long v) {
#if defined(__AMDGCN__)
    unsigned lo = (unsigned)__builtin_amdgcn_readfirstlane((int)(unsigned)v);
    unsigned hi = (unsigned)__builtin_amdgcn_readfirstlane((int)(unsigned)(v >> 32));
    return ((unsigned long long)hi << 32) | lo;
#else
    return v;
#endif
}
__device__ __forceinline__ unsigned rfl32(unsigned v) {
#if defined(__AMDGCN__)
    return (unsigned)__builtin_amdgcn_readfirstlane((int)v);
#else
    return v;
#endif
}

// ---------------- TDM: 2-D async tile  (32 k-elems) x (64 n-rows) of bf16 --------
// B is row-major [512][512] bf16 (row = output channel, k contiguous).
// D# per CDNA5 ISA. Each wave drives its own TDM chain;
// TENSORcnt is per-wave and TDM completes in order -> no cross-wave sync needed.
__device__ __forceinline__ void tdm_load_b(unsigned long long ga, unsigned lds_off) {
#if defined(__AMDGCN__)
    u32x4 g0 = { 1u,                                   // count=1 (valid), user mode
                 lds_off,                              // LDS byte offset (dyn base = 0)
                 (unsigned)ga,                         // global_addr[31:0]
                 (unsigned)(ga >> 32) | (2u << 30) };  // global_addr[56:32] | type=2
    i32x8 g1 = { (1 << 16),                            // wg_mask=0, data_size=1 (2 bytes)
                 (512 << 16),                          // tensor_dim0[15:0] (bits 63:48)
                 (512 << 16),                          // tensor_dim1[15:0] (bits 95:80)
                 (32 << 16),                           // tile_dim0 = 32    (bits 127:112)
                 64,                                   // tile_dim1 = 64, tile_dim2 = 0
                 512,                                  // tensor_dim0_stride[31:0]
                 0, 0 };
    i32x4 z4 = { 0, 0, 0, 0 };
    asm volatile("" ::: "memory");
#if __clang_major__ >= 23
    i32x8 z8 = { 0, 0, 0, 0, 0, 0, 0, 0 };
    __builtin_amdgcn_tensor_load_to_lds(g0, g1, z4, z4, z8, 0);
#else
    __builtin_amdgcn_tensor_load_to_lds(g0, g1, z4, z4, 0);
#endif
    asm volatile("" ::: "memory");
#else
    (void)ga; (void)lds_off;
#endif
}

__device__ __forceinline__ void wait_tensor(int n) {
#if defined(__AMDGCN__)
    if (n) __builtin_amdgcn_s_wait_tensorcnt(1);
    else   __builtin_amdgcn_s_wait_tensorcnt(0);
    asm volatile("" ::: "memory");
#else
    (void)n;
#endif
}

// ---------------- WMMA fragment helpers ----------------
// A/B 16-bit fragment layout (CDNA5 ISA): lanes 0-15 hold K {0..7, 16..23},
// lanes 16-31 hold K {8..15, 24..31} of their row -> two 16B loads.
__device__ __forceinline__ v16bf frag16(const bf16_t* p) {
    bf16x8 lo = *(const bf16x8*)(p);
    bf16x8 hi = *(const bf16x8*)(p + 16);
    return __builtin_shufflevector(lo, hi, 0,1,2,3,4,5,6,7,8,9,10,11,12,13,14,15);
}

// 16(M) x 64(N) x 512(K) per wave. A from global (WGP$-shared across the 8 waves,
// 1-step register prefetch); B streamed by this wave's private TDM double-buffer.
__device__ __forceinline__ void tile_gemm(const bf16_t* __restrict__ Aglob,
                                          const bf16_t* __restrict__ B,
                                          char* __restrict__ dynsmem,
                                          unsigned wave, int nwb, int l15, int lhi,
                                          v8f acc[4]) {
    const bf16_t* ap = Aglob + (size_t)l15 * DD + lhi * 8;
    // wave-uniform B base + LDS offsets, pinned into SGPRs so the per-chunk TDM
    // descriptor updates are pure SALU (no readfirstlane / VALU in the loop)
    const unsigned long long gb =
        rfl64((unsigned long long)(uintptr_t)(B + (size_t)nwb * DD));
    const unsigned off0 = rfl32(wave * 8192u), off1 = off0 + 4096u;
    bf16_t* lb0 = (bf16_t*)(dynsmem + off0);
    bf16_t* lb1 = (bf16_t*)(dynsmem + off1);

    tdm_load_b(gb, off0);                                // kt = 0 in flight
    v16bf a_cur = frag16(ap);                            // A(0) overlaps the DMA

#pragma unroll
    for (int kt = 0; kt < 16; ++kt) {
        const bf16_t* cb = (kt & 1) ? lb1 : lb0;
        if (kt < 15) tdm_load_b(gb + (unsigned long long)(kt + 1) * 64u,
                                (kt & 1) ? off0 : off1);
        wait_tensor(kt < 15 ? 1 : 0);                    // chunk kt ready, kt+1 in flight

        v16bf a_nxt = a_cur;
        if (kt < 15) a_nxt = frag16(ap + (kt + 1) * 32); // prefetch next A

        // gather all 4 B fragments first, then issue the 4 WMMAs
        v16bf bb[4];
#pragma unroll
        for (int t = 0; t < 4; ++t)
            bb[t] = frag16(cb + (size_t)(t * 16 + l15) * 32 + lhi * 8);
#pragma unroll
        for (int t = 0; t < 4; ++t)
            acc[t] = __builtin_amdgcn_wmma_f32_16x16x32_bf16(
                false, a_cur, false, bb[t], (short)0, acc[t], false, false);
        a_cur = a_nxt;
    }
}

// ---------------- peproj = pe @ Wpe^T + bias   (M = LSEQ) ----------------
__global__ __launch_bounds__(256) void gemm_pe(const bf16_t* __restrict__ peb,
                                               const bf16_t* __restrict__ Wpeb,
                                               const float* __restrict__ bias,
                                               float* __restrict__ peproj) {
    extern __shared__ char dynsmem[];                  // 64 KB: 8 waves x 2 x 4 KB

    const int tid  = threadIdx.x;
    const int wave = tid >> 5, lane = tid & 31;
    const int l15  = lane & 15, lhi = lane >> 4;
    const int Mbase = blockIdx.x * 16;                 // over LSEQ
    const int nwb   = wave * 64;

    v8f acc[4] = { v8f{0,0,0,0,0,0,0,0}, v8f{0,0,0,0,0,0,0,0},
                   v8f{0,0,0,0,0,0,0,0}, v8f{0,0,0,0,0,0,0,0} };
    tile_gemm(peb + (size_t)Mbase * DD, Wpeb, dynsmem, wave, nwb, l15, lhi, acc);

    // C/D layout: lanes 0-15 -> M=vgpr, lanes 16-31 -> M=vgpr+8; N = nwb+16t+l15
#pragma unroll
    for (int t = 0; t < 4; ++t) {
        int n = nwb + t * 16 + l15;
        float bn = bias[n];
#pragma unroll
        for (int v = 0; v < 8; ++v) {
            int l = Mbase + v + lhi * 8;
            peproj[(size_t)l * DD + n] = acc[t][v] + bn;
        }
    }
}

// ---------------- Newton solve of y^3/3 + y = x (nonsat activation) ----------------
__device__ __forceinline__ float nonsat(float x) {
    float y = x;
#pragma unroll
    for (int k = 0; k < 8; ++k) {
        float y2 = y * y;
        y = (0.66666667f * y2 * y + x) / (y2 + 1.0f);
    }
    return y;
}

// ---------------- main: out = LN(x @ Wx^T + peproj) -> nonsat ----------------
__global__ __launch_bounds__(256) void gemm_ln_act(const bf16_t* __restrict__ xb,
                                                   const bf16_t* __restrict__ Wxb,
                                                   const float* __restrict__ peproj,
                                                   const float* __restrict__ gamma,
                                                   const float* __restrict__ beta,
                                                   float* __restrict__ out) {
    extern __shared__ char dynsmem[];                  // 64 KB dynamic LDS
    float* tile = (float*)dynsmem;                     // epilogue: 16 x 512 f32 (32 KB)

    const int tid  = threadIdx.x;
    const int wave = tid >> 5, lane = tid & 31;
    const int l15  = lane & 15, lhi = lane >> 4;
    const int Mbase = blockIdx.x * 16;                 // over M_TOT rows
    const int nwb   = wave * 64;

    v8f acc[4] = { v8f{0,0,0,0,0,0,0,0}, v8f{0,0,0,0,0,0,0,0},
                   v8f{0,0,0,0,0,0,0,0}, v8f{0,0,0,0,0,0,0,0} };
    tile_gemm(xb + (size_t)Mbase * DD, Wxb, dynsmem, wave, nwb, l15, lhi, acc);

    __syncthreads();   // all waves done with their TDM buffers before tile overlays

    // dump accumulators through LDS to get row-contiguous access for LayerNorm
#pragma unroll
    for (int t = 0; t < 4; ++t)
#pragma unroll
        for (int v = 0; v < 8; ++v)
            tile[(v + lhi * 8) * DD + nwb + t * 16 + l15] = acc[t][v];
    __syncthreads();

    // 16 groups of 16 threads; group g owns row m = g of the tile
    const int g = tid >> 4, j = tid & 15;
    const int r = Mbase + g;                           // global row in (L*B)
    const int seq = r >> 3;                            // l = r / BATCH
    const float* pp = peproj + (size_t)seq * DD;

    float vals[32];
    float s = 0.f, sq = 0.f;
#pragma unroll
    for (int i = 0; i < 32; ++i) {
        int n = j + 16 * i;
        float v = tile[g * DD + n] + pp[n];
        vals[i] = v;
        s  += v;
        sq += v * v;
    }
    // reduce across the 16 lanes of this group (wave32-safe, width 16)
#pragma unroll
    for (int o = 8; o >= 1; o >>= 1) {
        s  += __shfl_xor(s,  o, 16);
        sq += __shfl_xor(sq, o, 16);
    }
    float mu  = s * (1.0f / DD);
    float var = sq * (1.0f / DD) - mu * mu;
    float rs  = rsqrtf(var + LN_EPS);

    float* orow = out + (size_t)r * DD;
#pragma unroll
    for (int i = 0; i < 32; ++i) {
        int n = j + 16 * i;
        float h = (vals[i] - mu) * rs * gamma[n] + beta[n];
        orow[n] = nonsat(h);
    }
}

// ---------------- launcher ----------------
extern "C" void kernel_launch(void* const* d_in, const int* in_sizes, int n_in,
                              void* d_out, int out_size, void* d_ws, size_t ws_size,
                              hipStream_t stream) {
    (void)in_sizes; (void)n_in; (void)out_size; (void)ws_size;
    const float* x     = (const float*)d_in[0];   // (L, B, D)
    const float* W     = (const float*)d_in[1];   // (D, 2D)
    const float* bias  = (const float*)d_in[2];   // (D,)
    const float* gamma = (const float*)d_in[3];   // (D,)
    const float* beta  = (const float*)d_in[4];   // (D,)
    const float* pe    = (const float*)d_in[5];   // (MAX_LEN, D)
    float* out = (float*)d_out;

    // workspace carve-up
    char* ws = (char*)d_ws;
    size_t off = 0;
    bf16_t* xb   = (bf16_t*)(ws + off); off += (size_t)M_TOT * DD * sizeof(bf16_t); // 32 MB
    bf16_t* Wxb  = (bf16_t*)(ws + off); off += (size_t)DD * DD * sizeof(bf16_t);    // 0.5 MB
    bf16_t* Wpeb = (bf16_t*)(ws + off); off += (size_t)DD * DD * sizeof(bf16_t);    // 0.5 MB
    bf16_t* peb  = (bf16_t*)(ws + off); off += (size_t)LSEQ * DD * sizeof(bf16_t);  // 4 MB
    float*  pprj = (float*) (ws + off);                                             // 8 MB

    // 1) conversions to bf16
    {
        int n4 = (M_TOT * DD) / 4;                       // x
        k_cvt<<<(n4 + 255) / 256, 256, 0, stream>>>(x, xb, n4);
    }
    {
        int n4 = (LSEQ * DD) / 4;                        // pe rows 0..LSEQ-1 (contiguous)
        k_cvt<<<(n4 + 255) / 256, 256, 0, stream>>>(pe, peb, n4);
    }
    {
        int n4 = (DD * DD) / 4;                          // W split
        k_cvt_w<<<(n4 + 255) / 256, 256, 0, stream>>>(W, Wxb, Wpeb);
    }

    // 2) positional projection: peproj = pe @ Wpe^T + b   (shared by all batch rows)
    gemm_pe<<<LSEQ / 16, 256, 64 * 1024, stream>>>(peb, Wpeb, bias, pprj);

    // 3) main fused GEMM + LayerNorm + Newton activation
    gemm_ln_act<<<M_TOT / 16, 256, 64 * 1024, stream>>>(xb, Wxb, pprj, gamma, beta, out);
}
